// QuantumRBFKernel_65481071397084
// MI455X (gfx1250) — matrix operand order repaired and checked by reading, not compiled, not run on hardware
//
#include <hip/hip_runtime.h>
#include <hip/hip_bf16.h>

typedef _Float16 v16h __attribute__((ext_vector_type(16)));
typedef _Float16 v8h  __attribute__((ext_vector_type(8)));
typedef float    v8f  __attribute__((ext_vector_type(8)));
typedef int      v4i  __attribute__((vector_size(16)));   // matches builtin param type

typedef __attribute__((address_space(1))) v4i* gv4i_ptr;  // global int4*
typedef __attribute__((address_space(3))) v4i* lv4i_ptr;  // LDS int4*

#define NROWS 8192
#define MCOLS 8192
#define DDIM  512
#define BM 128
#define BN 128
#define BK 32
#define LDSS 40          // halves per LDS row: 32 data + 8 pad -> 80B stride (bank-conflict free, 16B aligned)
#define GAMMA_F 0.002f

// CDNA5 async global->LDS copy (ASYNCcnt-tracked). Guarded: falls back to
// VGPR-staged copies if the toolchain lacks the gfx1250 builtins.
#if __has_builtin(__builtin_amdgcn_global_load_async_to_lds_b128) && \
    __has_builtin(__builtin_amdgcn_s_wait_asynccnt)
#define USE_ASYNC 1
#else
#define USE_ASYNC 0
#endif

#if USE_ASYNC
#define ASYNC_CP(ldst, gsrc)                                                   \
  __builtin_amdgcn_global_load_async_to_lds_b128(                              \
      (gv4i_ptr)(void*)const_cast<_Float16*>(gsrc),                            \
      (lv4i_ptr)(void*)(ldst), 0, 0)
#define WAIT_ASYNC(n) __builtin_amdgcn_s_wait_asynccnt(n)
#else
#define ASYNC_CP(ldst, gsrc) (*(v8h*)(ldst) = *(const v8h*)(gsrc))
#define WAIT_ASYNC(n) ((void)0)
#endif

// ---------------------------------------------------------------------------
// Pass 1: per-row sum of squares (fp32) + hi/lo f16 split of each element.
// x ~= hi + lo, hi = f16(x), lo = f16(x - hi)  -> ~22 mantissa bits recovered.
// ---------------------------------------------------------------------------
__global__ __launch_bounds__(256) void rbf_prep(const float* __restrict__ src,
                                                _Float16* __restrict__ hi,
                                                _Float16* __restrict__ lo,
                                                float* __restrict__ sq) {
  __shared__ float red[256];
  const int row = blockIdx.x;
  const int t = threadIdx.x;
  const float* r = src + (size_t)row * DDIM;
  float s = 0.f;
#pragma unroll
  for (int i = t; i < DDIM; i += 256) {
    float v = r[i];
    _Float16 h = (_Float16)v;
    float l = v - (float)h;
    hi[(size_t)row * DDIM + i] = h;
    lo[(size_t)row * DDIM + i] = (_Float16)l;
    s += v * v;
  }
  red[t] = s;
  __syncthreads();
  for (int off = 128; off > 0; off >>= 1) {
    if (t < off) red[t] += red[t + off];
    __syncthreads();
  }
  if (t == 0) sq[row] = red[0];
}

// ---------------------------------------------------------------------------
// Pass 2: fused (split-f16 WMMA GEMM) + RBF epilogue.
// Block = 256 threads (8 wave32), tile = 128x128, BK = 32, double-buffered LDS.
// Wave (wm,wn) = (wave>>1, wave&1) owns a 32x64 subtile = 2x4 WMMA tiles.
// Three WMMAs per tile per K-step: hi*hi + hi*lo + lo*hi, fp32 accumulate.
// Pipeline: barrier -> async-load next tile -> s_wait_asynccnt 8 (prev tile
// landed, in-order) -> barrier -> 24x WMMA on current tile.
// ---------------------------------------------------------------------------
__global__ __launch_bounds__(256) void rbf_main(const _Float16* __restrict__ xh,
                                                const _Float16* __restrict__ xl,
                                                const _Float16* __restrict__ yh,
                                                const _Float16* __restrict__ yl,
                                                const float* __restrict__ xsq,
                                                const float* __restrict__ ysq,
                                                float* __restrict__ out) {
  __shared__ _Float16 sXh[2][BM * LDSS];
  __shared__ _Float16 sXl[2][BM * LDSS];
  __shared__ _Float16 sYh[2][BN * LDSS];
  __shared__ _Float16 sYl[2][BN * LDSS];
  __shared__ float sXs[BM];
  __shared__ float sYs[BN];

  const int tid  = threadIdx.x;
  const int lane = tid & 31;
  const int wave = tid >> 5;
  const int wm = wave >> 1;   // 0..3 : 32-row band
  const int wn = wave & 1;    // 0..1 : 64-col band
  const int block_row = blockIdx.y * BM;
  const int block_col = blockIdx.x * BN;

  if (tid < BM) sXs[tid] = xsq[block_row + tid];
  else          sYs[tid - BM] = ysq[block_col + (tid - BM)];

  // Per-tile copy: 4 planes x 128x32 halves; 512 16B-chunks per plane,
  // 2 chunks per thread -> 8 async b128 ops per wave per tile.
  auto tile_load = [&](int buf, int k0) {
#pragma unroll
    for (int it = 0; it < 2; ++it) {
      int c   = tid + it * 256;
      int row = c >> 2;
      int c4  = (c & 3) << 3;                      // half offset: 0,8,16,24
      size_t gx = (size_t)(block_row + row) * DDIM + k0 + c4;
      size_t gy = (size_t)(block_col + row) * DDIM + k0 + c4;
      int l = row * LDSS + c4;
      ASYNC_CP(&sXh[buf][l], xh + gx);
      ASYNC_CP(&sXl[buf][l], xl + gx);
      ASYNC_CP(&sYh[buf][l], yh + gy);
      ASYNC_CP(&sYl[buf][l], yl + gy);
    }
  };

  // ISA 7.12.2 fragment addressing (wave32):
  // A (16x32 f16): lane holds M = lane%16, K chunks {koffA..+7, koffA+16..+23}, koffA = (lane>=16)*8
  // B (32x16 f16): lane holds N = lane%16, K chunks {koffB..+7, koffB+16..+23}, koffB = (lane>=16)*8
  const int frow  = lane & 15;
  const int koffA = (lane >> 4) << 3;
  const int koffB = (lane >> 4) << 3;

  v8f acc[2][4] = {};

  tile_load(0, 0);  // prologue: stream first tile into buffer 0
  int cur = 0;

  for (int k0 = 0; k0 < DDIM; k0 += BK) {
    __syncthreads();  // all waves done computing from buffer cur^1
    if (k0 + BK < DDIM) {
      tile_load(cur ^ 1, k0 + BK);  // overlap next-tile copy with compute
      WAIT_ASYNC(8);                // current tile's 8 async ops complete
    } else {
      WAIT_ASYNC(0);
    }
    __syncthreads();  // cross-wave: current buffer fully populated

    // B fragments for the wave's 4 column tiles (hi + lo planes).
    v16h bh[4], bl[4];
#pragma unroll
    for (int tn = 0; tn < 4; ++tn) {
      int rB = (wn * 64 + tn * 16 + frow) * LDSS + koffB;
      union { v16h v; v8h h[2]; } ubh, ubl;
      ubh.h[0] = *(const v8h*)&sYh[cur][rB];
      ubh.h[1] = *(const v8h*)&sYh[cur][rB + 16];
      ubl.h[0] = *(const v8h*)&sYl[cur][rB];
      ubl.h[1] = *(const v8h*)&sYl[cur][rB + 16];
      bh[tn] = ubh.v;
      bl[tn] = ubl.v;
    }

#pragma unroll
    for (int tm = 0; tm < 2; ++tm) {
      int rA = (wm * 32 + tm * 16 + frow) * LDSS + koffA;
      union { v16h v; v8h h[2]; } uah, ual;
      uah.h[0] = *(const v8h*)&sXh[cur][rA];
      uah.h[1] = *(const v8h*)&sXh[cur][rA + 16];
      ual.h[0] = *(const v8h*)&sXl[cur][rA];
      ual.h[1] = *(const v8h*)&sXl[cur][rA + 16];
      v16h ah = uah.v, al = ual.v;
#pragma unroll
      for (int tn = 0; tn < 4; ++tn) {
        acc[tm][tn] = __builtin_amdgcn_wmma_f32_16x16x32_f16(
            false, ah, false, bh[tn], (short)0, acc[tm][tn], false, false);
        acc[tm][tn] = __builtin_amdgcn_wmma_f32_16x16x32_f16(
            false, ah, false, bl[tn], (short)0, acc[tm][tn], false, false);
        acc[tm][tn] = __builtin_amdgcn_wmma_f32_16x16x32_f16(
            false, al, false, bh[tn], (short)0, acc[tm][tn], false, false);
      }
    }
    cur ^= 1;
  }

  // Fused RBF epilogue. C/D layout: lane L, VGPR r -> M = r + 8*(L>=16), N = L%16.
  const int rhalf = (lane >> 4) << 3;
#pragma unroll
  for (int tm = 0; tm < 2; ++tm) {
#pragma unroll
    for (int tn = 0; tn < 4; ++tn) {
      int colLocal = wn * 64 + tn * 16 + (lane & 15);
      int col = block_col + colLocal;
      float ys = sYs[colLocal];
#pragma unroll
      for (int r = 0; r < 8; ++r) {
        int rowLocal = wm * 32 + tm * 16 + r + rhalf;
        int row = block_row + rowLocal;
        float sd = sXs[rowLocal] + ys - 2.0f * acc[tm][tn][r];
        sd = fmaxf(sd, 0.0f);
        out[(size_t)row * MCOLS + col] = __expf(-GAMMA_F * sd);
      }
    }
  }
}

// ---------------------------------------------------------------------------
// Host-side launcher
// ---------------------------------------------------------------------------
extern "C" void kernel_launch(void* const* d_in, const int* in_sizes, int n_in,
                              void* d_out, int out_size, void* d_ws, size_t ws_size,
                              hipStream_t stream) {
  const float* x = (const float*)d_in[0];
  const float* y = (const float*)d_in[1];
  float* out = (float*)d_out;

  // Workspace layout (~32.06 MB):
  // [x_hi | x_lo | y_hi | y_lo] f16 planes, then x_sq, y_sq fp32 vectors.
  _Float16* xh = (_Float16*)d_ws;
  _Float16* xl = xh + (size_t)NROWS * DDIM;
  _Float16* yh = xl + (size_t)NROWS * DDIM;
  _Float16* yl = yh + (size_t)MCOLS * DDIM;
  float* xsq = (float*)(yl + (size_t)MCOLS * DDIM);
  float* ysq = xsq + NROWS;

  rbf_prep<<<NROWS, 256, 0, stream>>>(x, xh, xl, xsq);
  rbf_prep<<<MCOLS, 256, 0, stream>>>(y, yh, yl, ysq);

  dim3 grid(MCOLS / BN, NROWS / BM);
  rbf_main<<<grid, 256, 0, stream>>>(xh, xl, yh, yl, xsq, ysq, out);
}
